// STLGTransformerBlock_23441931501669
// MI455X (gfx1250) — matrix operand order repaired and checked by reading, not compiled, not run on hardware
//
#include <hip/hip_runtime.h>
#include <hip/hip_fp16.h>

// ---------------------------------------------------------------------------
// MI455X (gfx1250) implementation of the STLG transformer block.
// Convolutions are implicit GEMMs on v_wmma_f32_16x16x32_f16 (wave32):
//  - 2 out-channel tiles per wave (2 WMMAs per A fragment)
//  - depth-1 software pipeline over K chunks (overlap VMEM with matrix pipe)
//  - zero-page for "same" padding -> branch-free unconditional loads
// Attention (~2% of FLOPs) uses VALU online-softmax with LDS-staged K/V.
// ---------------------------------------------------------------------------

typedef __attribute__((ext_vector_type(16))) _Float16 v16h;
typedef __attribute__((ext_vector_type(8)))  _Float16 v8h;
typedef __attribute__((ext_vector_type(8)))  float    v8f;

#define NTOK 16384          // 4*16*16*16 tokens
#define CDIM 128
#define NKV  1032           // 8 coarse + 1024 fine kv tokens

// ---------------------------------------------------------------------------
// Weight packing: w[Cout][Cin][taps] (f32) -> WMMA-B-friendly f16 layout:
//   wp[ (((tap*nC + cc)*nO + ot)*32 + lane)*16 + e ]
// B-operand layout (16-bit B 32x16): lane l, element e holds
//   K = 8*(l>>4) + (e&7) + 16*(e>>3), column N = l&15
// ---------------------------------------------------------------------------
__global__ void pack_w_kernel(const float* __restrict__ w, _Float16* __restrict__ wp,
                              int Cin, int Cout, int taps) {
  const int idx = blockIdx.x * blockDim.x + threadIdx.x;
  const int total = taps * Cin * Cout;
  if (idx >= total) return;
  const int e    = idx & 15;
  const int lane = (idx >> 4) & 31;
  int rest       = idx >> 9;
  const int nO = Cout >> 4;
  const int nC = Cin >> 5;
  const int ot = rest % nO; rest /= nO;
  const int cc = rest % nC;
  const int tap = rest / nC;
  const int col = lane & 15;
  const int k   = ((lane >> 4) << 3) + (e & 7) + ((e >> 3) << 4);   // B layout: K = 8*hi + (e&7) + 16*(e>>3)
  const int ic  = cc * 32 + k;
  const int oc  = ot * 16 + col;
  wp[idx] = (_Float16)w[((size_t)oc * Cin + ic) * (size_t)taps + tap];
}

__global__ void zero_h_kernel(_Float16* __restrict__ p, int n) {
  const int i = blockIdx.x * blockDim.x + threadIdx.x;
  if (i < n) p[i] = (_Float16)0.0f;
}

// ---------------------------------------------------------------------------
// Implicit-GEMM 4D convolution ("same" padding, arbitrary stride) with WMMA.
// One wave computes a 16-token x 32-outchannel tile (two 16x16 WMMA tiles).
// K loop = TAPS * CIN/32, software-pipelined depth 1.
// X:  [Nin , CIN ] f16 token-major.   Y/Yh: [Nout, COUT] f32 / f16.
// R:  optional residual [Nout, COUT] f32 added after bias (+relu).
// Zp: zero page (>= CIN halves of 0) for padded taps / tail rows.
// ---------------------------------------------------------------------------
template<int KL, int KD, int KH, int KW, int CIN, int COUT>
__global__ __launch_bounds__(256) void conv4d_wmma_kernel(
    const _Float16* __restrict__ X, const _Float16* __restrict__ Wp,
    const float* __restrict__ bias, const float* __restrict__ R,
    float* __restrict__ Y, _Float16* __restrict__ Yh,
    const _Float16* __restrict__ Zp,
    int iL, int iD, int iH, int iW, int oL, int oD, int oH, int oW,
    int sl, int sd, int sh, int sw, int relu)
{
  constexpr int NC   = CIN  / 32;   // K chunks of 32
  constexpr int NO   = COUT / 16;   // 16-wide oc tiles
  constexpr int NO2  = NO / 2;      // oc tile *pairs* per token tile
  constexpr int TAPS = KL * KD * KH * KW;

  const int Nout = oL * oD * oH * oW;
  const int tokenTiles = (Nout + 15) >> 4;
  const int wave = blockIdx.x * (blockDim.x >> 5) + (threadIdx.x >> 5);
  if (wave >= tokenTiles * NO2) return;          // wave-uniform exit
  const int og = (wave % NO2) * 2;               // first oc tile of the pair
  const int tt = wave / NO2;
  const int lane = threadIdx.x & 31;
  const int hi   = lane >> 4;                    // half-wave id
  const int kbA  = hi << 3;                      // A-matrix K base (0 or 8)
  const int row  = lane & 15;                    // token row of the tile
  const int tok  = tt * 16 + row;
  const bool tokValid = tok < Nout;

  int ol = 0, od = 0, oh = 0, ow = 0;
  if (tokValid) { int t = tok; ow = t % oW; t /= oW; oh = t % oH; t /= oH; od = t % oD; ol = t / oD; }

  v8f acc0, acc1;
#pragma unroll
  for (int i = 0; i < 8; ++i) { acc0[i] = 0.0f; acc1[i] = 0.0f; }

  union AV { v16h v; v8h h[2]; };

  for (int tap = 0; tap < TAPS; ++tap) {
    // tap geometry: all div/mod by compile-time constants
    const int fw = tap % KW;
    const int fh = (tap / KW) % KH;
    const int fd = (tap / (KW * KH)) % KD;
    const int fl = tap / (KW * KH * KD);
    const int il = ol * sl + fl - (KL >> 1);
    const int id = od * sd + fd - (KD >> 1);
    const int ih = oh * sh + fh - (KH >> 1);
    const int iw = ow * sw + fw - (KW >> 1);
    const bool valid = tokValid &&
        (unsigned)il < (unsigned)iL && (unsigned)id < (unsigned)iD &&
        (unsigned)ih < (unsigned)iH && (unsigned)iw < (unsigned)iW;
    const size_t src = ((((size_t)il * iD + id) * iH + ih) * iW + iw);
    // zero-page: invalid lanes read zeros, loads stay unconditional
    const _Float16* xrow = valid ? (X + src * (size_t)CIN) : Zp;
    const _Float16* w0 = Wp + ((((size_t)tap * NC) * NO + og) * 32 + lane) * 16;
    // prefetch next tap's weights (global_prefetch_b8)
    __builtin_prefetch(w0 + (size_t)NC * NO * 512, 0, 0);

    // depth-1 software pipeline over the NC chunks
    AV   a[2];
    v16h b0[2], b1[2];
    {
      const v8h* p = reinterpret_cast<const v8h*>(xrow + kbA);
      a[0].h[0] = p[0];                 // K = kbA + 0..7
      a[0].h[1] = p[2];                 // K = 16 + kbA + 0..7
      b0[0] = *reinterpret_cast<const v16h*>(w0);
      b1[0] = *reinterpret_cast<const v16h*>(w0 + 512);
    }
#pragma unroll
    for (int cc = 0; cc < NC; ++cc) {
      const int cur = cc & 1;
      const int nxt = cur ^ 1;
      if (cc + 1 < NC) {
        const v8h* p = reinterpret_cast<const v8h*>(xrow + (cc + 1) * 32 + kbA);
        a[nxt].h[0] = p[0];
        a[nxt].h[1] = p[2];
        const _Float16* wc = w0 + (size_t)(cc + 1) * NO * 512;
        b0[nxt] = *reinterpret_cast<const v16h*>(wc);
        b1[nxt] = *reinterpret_cast<const v16h*>(wc + 512);
      }
      acc0 = __builtin_amdgcn_wmma_f32_16x16x32_f16(false, a[cur].v, false, b0[cur],
                                                    (short)0, acc0, false, false);
      acc1 = __builtin_amdgcn_wmma_f32_16x16x32_f16(false, a[cur].v, false, b1[cur],
                                                    (short)0, acc1, false, false);
    }
  }

  // Epilogue: C/D layout — VGPR i holds (M = i + 8*hi, N = lane&15)
  const int col = lane & 15;
  const float bv0 = bias[og * 16 + col];
  const float bv1 = bias[og * 16 + 16 + col];
  const int mbase = tt * 16 + (hi << 3);
#pragma unroll
  for (int i = 0; i < 8; ++i) {
    const int m = mbase + i;
    if (m < Nout) {
      float v0 = acc0[i] + bv0;
      float v1 = acc1[i] + bv1;
      if (relu) { v0 = v0 > 0.f ? v0 : 0.f; v1 = v1 > 0.f ? v1 : 0.f; }
      const size_t off0 = (size_t)m * COUT + og * 16 + col;
      const size_t off1 = off0 + 16;
      if (R)  { v0 += R[off0]; v1 += R[off1]; }
      if (Y)  { Y[off0]  = v0;             Y[off1]  = v1; }
      if (Yh) { Yh[off0] = (_Float16)v0;   Yh[off1] = (_Float16)v1; }
    }
  }
}

template<int KL, int KD, int KH, int KW, int CI, int CO>
static void launch_conv(hipStream_t stream,
    const _Float16* X, const _Float16* Wp, const float* bias,
    const float* R, float* Y, _Float16* Yh, const _Float16* zp,
    int iL, int iD, int iH, int iW, int oL, int oD, int oH, int oW,
    int sl, int sd, int sh, int sw, int relu)
{
  const int Nout = oL * oD * oH * oW;
  const int tiles = ((Nout + 15) / 16) * (CO / 32);   // oc tile pairs
  const int blocks = (tiles + 7) / 8;                 // 8 waves / block
  conv4d_wmma_kernel<KL, KD, KH, KW, CI, CO><<<blocks, 256, 0, stream>>>(
      X, Wp, bias, R, Y, Yh, zp,
      iL, iD, iH, iW, oL, oD, oH, oW, sl, sd, sh, sw, relu);
}

// ---------------------------------------------------------------------------
// Channel LayerNorm over C=128 per token; one wave per token (wave32: 4 ch/lane)
// ---------------------------------------------------------------------------
__global__ __launch_bounds__(256) void cnorm_kernel(
    const float* __restrict__ X, const float* __restrict__ w, const float* __restrict__ b,
    float* __restrict__ Y, _Float16* __restrict__ Yh, int N)
{
  const int wv = blockIdx.x * (blockDim.x >> 5) + (threadIdx.x >> 5);
  if (wv >= N) return;
  const int lane = threadIdx.x & 31;
  const size_t base = (size_t)wv * 128 + lane * 4;
  float xv[4];
#pragma unroll
  for (int j = 0; j < 4; ++j) xv[j] = X[base + j];
  float s = 0.f, s2 = 0.f;
#pragma unroll
  for (int j = 0; j < 4; ++j) { s += xv[j]; s2 += xv[j] * xv[j]; }
  for (int off = 16; off > 0; off >>= 1) {       // wave32 butterfly reduction
    s  += __shfl_xor(s,  off, 32);
    s2 += __shfl_xor(s2, off, 32);
  }
  const float mu  = s  * (1.f / 128.f);
  const float var = s2 * (1.f / 128.f) - mu * mu;
  const float inv = rsqrtf(var + 1e-5f);
#pragma unroll
  for (int j = 0; j < 4; ++j) {
    const int c = lane * 4 + j;
    const float o = (xv[j] - mu) * inv * w[c] + b[c];
    if (Y)  Y[base + j]  = o;
    if (Yh) Yh[base + j] = (_Float16)o;
  }
}

// ---------------------------------------------------------------------------
// tlg windowed cross-attention: LOCAL=(1,2,2,2) -> 8 q tokens vs 8 kv tokens,
// 8 heads, head_dim 16. One block (64 thr) per window; thread = (head, qpos).
// Grids are per-L-half: l in [0,2), token = ((l*16+d)*16+h)*16+w.
// ---------------------------------------------------------------------------
__device__ __forceinline__ int win_token(int nl_, int nd_, int nh_, int nw_, int p) {
  const int dd = (p >> 2) & 1, dh = (p >> 1) & 1, dw = p & 1;
  const int l = nl_, d = 2 * nd_ + dd, h = 2 * nh_ + dh, w = 2 * nw_ + dw;
  return ((l * 16 + d) * 16 + h) * 16 + w;
}

__global__ __launch_bounds__(64) void win_attn_kernel(
    const float* __restrict__ Q, const float* __restrict__ K,
    const float* __restrict__ V, _Float16* __restrict__ Yh)
{
  __shared__ float ks[8 * 128];
  __shared__ float vs[8 * 128];
  const int wid = blockIdx.x;
  const int nw_ = wid & 7, nh_ = (wid >> 3) & 7, nd_ = (wid >> 6) & 7, nl_ = wid >> 9;
  const int tid = threadIdx.x;

  for (int i = tid; i < 1024; i += 64) {
    const int p = i >> 7, c = i & 127;
    const int t = win_token(nl_, nd_, nh_, nw_, p);
    ks[i] = K[(size_t)t * 128 + c];
    vs[i] = V[(size_t)t * 128 + c];
  }
  __syncthreads();

  const int qp = tid & 7, head = tid >> 3;
  const int qt = win_token(nl_, nd_, nh_, nw_, qp);
  float qv[16];
#pragma unroll
  for (int i = 0; i < 16; ++i) qv[i] = Q[(size_t)qt * 128 + head * 16 + i];

  float sc[8]; float mx = -1e30f;
#pragma unroll
  for (int j = 0; j < 8; ++j) {
    float s = 0.f;
#pragma unroll
    for (int i = 0; i < 16; ++i) s += qv[i] * ks[j * 128 + head * 16 + i];
    s *= 0.25f;                                  // SCALE = 16^-0.5
    sc[j] = s; mx = fmaxf(mx, s);
  }
  float sum = 0.f;
#pragma unroll
  for (int j = 0; j < 8; ++j) { sc[j] = __expf(sc[j] - mx); sum += sc[j]; }
  const float inv = 1.f / sum;
#pragma unroll
  for (int i = 0; i < 16; ++i) {
    float y = 0.f;
#pragma unroll
    for (int j = 0; j < 8; ++j) y += sc[j] * vs[j * 128 + head * 16 + i];
    Yh[(size_t)qt * 128 + head * 16 + i] = (_Float16)(y * inv);
  }
}

// ---------------------------------------------------------------------------
// slg global attention: 16384 q tokens vs Nk=1032 kv, 8 heads, head_dim 16.
// Online softmax; K/V chunked through LDS (128 keys/chunk, 8KB total).
// Grid: (64, heads); block 256; one q token per thread.
// ---------------------------------------------------------------------------
__global__ __launch_bounds__(256) void global_attn_kernel(
    const float* __restrict__ Q, const _Float16* __restrict__ Kh,
    const _Float16* __restrict__ Vh, _Float16* __restrict__ Yh, int Nk)
{
  __shared__ _Float16 ks[128 * 16];
  __shared__ _Float16 vsm[128 * 16];
  const int tid  = threadIdx.x;
  const int head = blockIdx.y;
  const int q    = blockIdx.x * 256 + tid;

  float qv[16];
#pragma unroll
  for (int i = 0; i < 16; ++i) qv[i] = Q[(size_t)q * 128 + head * 16 + i];
  float M = -1e30f, S = 0.f, acc[16];
#pragma unroll
  for (int i = 0; i < 16; ++i) acc[i] = 0.f;

  for (int c0 = 0; c0 < Nk; c0 += 128) {
    const int cnt = (Nk - c0 < 128) ? (Nk - c0) : 128;
    __syncthreads();
    for (int i = tid; i < cnt * 16; i += 256) {
      const int j = i >> 4, ii = i & 15;
      ks[i]  = Kh[(size_t)(c0 + j) * 128 + head * 16 + ii];
      vsm[i] = Vh[(size_t)(c0 + j) * 128 + head * 16 + ii];
    }
    __syncthreads();
    for (int j = 0; j < cnt; ++j) {
      float s = 0.f;
#pragma unroll
      for (int i = 0; i < 16; ++i) s += qv[i] * (float)ks[j * 16 + i];
      s *= 0.25f;
      const float nM = fmaxf(M, s);
      const float rs = __expf(M - nM);
      const float p  = __expf(s - nM);
      S = S * rs + p;
#pragma unroll
      for (int i = 0; i < 16; ++i) acc[i] = acc[i] * rs + p * (float)vsm[j * 16 + i];
      M = nM;
    }
  }
  const float inv = 1.f / S;
#pragma unroll
  for (int i = 0; i < 16; ++i)
    Yh[(size_t)q * 128 + head * 16 + i] = (_Float16)(acc[i] * inv);
}

// ---------------------------------------------------------------------------
// Layout shuffles: (B,L,C,D,H,W) <-> token-major [N=L*D*H*W, C]
// ---------------------------------------------------------------------------
__global__ void x_to_tok_kernel(const float* __restrict__ X, float* __restrict__ T) {
  const int i = blockIdx.x * blockDim.x + threadIdx.x;
  if (i >= NTOK * CDIM) return;
  const int c = i & 127; const int n = i >> 7;
  const int w = n & 15, h = (n >> 4) & 15, d = (n >> 8) & 15, l = n >> 12;
  T[i] = X[((((size_t)l * 128 + c) * 16 + d) * 16 + h) * 16 + w];
}

__global__ void tok_to_x_kernel(const float* __restrict__ T, float* __restrict__ X) {
  const int i = blockIdx.x * blockDim.x + threadIdx.x;
  if (i >= NTOK * CDIM) return;
  const int c = i & 127; const int n = i >> 7;
  const int w = n & 15, h = (n >> 4) & 15, d = (n >> 8) & 15, l = n >> 12;
  X[((((size_t)l * 128 + c) * 16 + d) * 16 + h) * 16 + w] = T[i];
}

// ---------------------------------------------------------------------------
// Host orchestration
// ---------------------------------------------------------------------------
extern "C" void kernel_launch(void* const* d_in, const int* in_sizes, int n_in,
                              void* d_out, int out_size, void* d_ws, size_t ws_size,
                              hipStream_t stream) {
  (void)in_sizes; (void)n_in; (void)out_size; (void)ws_size;

  const float* x   = (const float*)d_in[0];
  const float* n1w = (const float*)d_in[1],  *n1b = (const float*)d_in[2];
  const float* n2w = (const float*)d_in[3],  *n2b = (const float*)d_in[4];
  const float* n3w = (const float*)d_in[5],  *n3b = (const float*)d_in[6];
  const float* snw = (const float*)d_in[7],  *snb = (const float*)d_in[8];
  const float* tqw = (const float*)d_in[9],  *tqb = (const float*)d_in[10];
  const float* tkw = (const float*)d_in[11], *tkb = (const float*)d_in[12];
  const float* tvw = (const float*)d_in[13], *tvb = (const float*)d_in[14];
  const float* tpw = (const float*)d_in[15], *tpb = (const float*)d_in[16];
  const float* sqw = (const float*)d_in[17], *sqb = (const float*)d_in[18];
  const float* skw = (const float*)d_in[19], *skb = (const float*)d_in[20];
  const float* svw = (const float*)d_in[21], *svb = (const float*)d_in[22];
  const float* spw = (const float*)d_in[23], *spb = (const float*)d_in[24];
  const float* sfw = (const float*)d_in[25], *sfb = (const float*)d_in[26];
  const float* scw = (const float*)d_in[27], *scb = (const float*)d_in[28];
  const float* m1w = (const float*)d_in[29], *m1b = (const float*)d_in[30];
  const float* m2w = (const float*)d_in[31], *m2b = (const float*)d_in[32];

  const size_t NF = (size_t)NTOK * CDIM;       // 2,097,152

  char* wsp = (char*)d_ws;
  auto alloc = [&](size_t bytes) -> void* {
    void* r = (void*)wsp; wsp += (bytes + 255) & ~(size_t)255; return r;
  };

  float*    bufA = (float*)alloc(NF * 4);      // residual stream / final
  float*    bufB = (float*)alloc(NF * 4);
  float*    bufC = (float*)alloc(NF * 4);
  _Float16* hH   = (_Float16*)alloc(NF * 2);   // normed activations (f16)
  float*    qF   = (float*)alloc(NF * 4);
  float*    kF   = (float*)alloc(NF * 4);
  float*    vF   = (float*)alloc(NF * 4);
  _Float16* yH   = (_Float16*)alloc(NF * 2);   // attention outputs (f16)
  _Float16* xcH  = (_Float16*)alloc((size_t)8 * 128 * 2);
  _Float16* xfH  = (_Float16*)alloc((size_t)1024 * 128 * 2);
  float*    kvkF = (float*)alloc((size_t)NKV * 128 * 4);
  float*    kvvF = (float*)alloc((size_t)NKV * 128 * 4);
  _Float16* knH  = (_Float16*)alloc((size_t)NKV * 128 * 2);
  _Float16* vnH  = (_Float16*)alloc((size_t)NKV * 128 * 2);
  _Float16* midH = (_Float16*)alloc((size_t)NTOK * 512 * 2);
  _Float16* zp   = (_Float16*)alloc((size_t)512 * 2);   // zero page (max CIN)

  _Float16* p_tq = (_Float16*)alloc((size_t)3 * 128 * 128 * 2);
  _Float16* p_tk = (_Float16*)alloc((size_t)3 * 128 * 128 * 2);
  _Float16* p_tv = (_Float16*)alloc((size_t)3 * 128 * 128 * 2);
  _Float16* p_tp = (_Float16*)alloc((size_t)3 * 128 * 128 * 2);
  _Float16* p_sq = (_Float16*)alloc((size_t)27 * 128 * 128 * 2);
  _Float16* p_sk = (_Float16*)alloc((size_t)27 * 128 * 128 * 2);
  _Float16* p_sv = (_Float16*)alloc((size_t)27 * 128 * 128 * 2);
  _Float16* p_sp = (_Float16*)alloc((size_t)27 * 128 * 128 * 2);
  _Float16* p_sf = (_Float16*)alloc((size_t)27 * 128 * 128 * 2);
  _Float16* p_sc = (_Float16*)alloc((size_t)27 * 128 * 128 * 2);
  _Float16* p_m1 = (_Float16*)alloc((size_t)81 * 128 * 512 * 2);
  _Float16* p_m2 = (_Float16*)alloc((size_t)81 * 512 * 128 * 2);

  auto pack = [&](const float* w, _Float16* wp, int Cin, int Cout, int taps) {
    const int total = taps * Cin * Cout;
    pack_w_kernel<<<(total + 255) / 256, 256, 0, stream>>>(w, wp, Cin, Cout, taps);
  };
  auto cnorm = [&](const float* X, const float* w, const float* b,
                   float* Y, _Float16* Yh, int N) {
    cnorm_kernel<<<(N + 7) / 8, 256, 0, stream>>>(X, w, b, Y, Yh, N);
  };

  // ---- zero page + weight packing (f32 -> WMMA-lane-ordered f16) ----
  zero_h_kernel<<<2, 256, 0, stream>>>(zp, 512);
  pack(tqw, p_tq, 128, 128, 3);  pack(tkw, p_tk, 128, 128, 3);
  pack(tvw, p_tv, 128, 128, 3);  pack(tpw, p_tp, 128, 128, 3);
  pack(sqw, p_sq, 128, 128, 27); pack(skw, p_sk, 128, 128, 27);
  pack(svw, p_sv, 128, 128, 27); pack(spw, p_sp, 128, 128, 27);
  pack(sfw, p_sf, 128, 128, 27); pack(scw, p_sc, 128, 128, 27);
  pack(m1w, p_m1, 128, 512, 81); pack(m2w, p_m2, 512, 128, 81);

  // ---- input -> token-major residual stream ----
  x_to_tok_kernel<<<(int)((NF + 255) / 256), 256, 0, stream>>>(x, bufA);

  const size_t HOFF = (size_t)8192 * 128;          // L-half offset (elements)

  // ================= Stage 1: tlg (temporal windowed cross-attention) =======
  cnorm(bufA, n1w, n1b, nullptr, hH, NTOK);
  // q/k/v convs (3,1,1,1) on each L-half (L=2 grid, zero-padded at half edges)
  launch_conv<3,1,1,1,128,128>(stream, hH,        p_tq, tqb, nullptr, qF,        nullptr, zp, 2,16,16,16, 2,16,16,16, 1,1,1,1, 0);
  launch_conv<3,1,1,1,128,128>(stream, hH + HOFF, p_tq, tqb, nullptr, qF + HOFF, nullptr, zp, 2,16,16,16, 2,16,16,16, 1,1,1,1, 0);
  launch_conv<3,1,1,1,128,128>(stream, hH,        p_tk, tkb, nullptr, kF,        nullptr, zp, 2,16,16,16, 2,16,16,16, 1,1,1,1, 0);
  launch_conv<3,1,1,1,128,128>(stream, hH + HOFF, p_tk, tkb, nullptr, kF + HOFF, nullptr, zp, 2,16,16,16, 2,16,16,16, 1,1,1,1, 0);
  launch_conv<3,1,1,1,128,128>(stream, hH,        p_tv, tvb, nullptr, vF,        nullptr, zp, 2,16,16,16, 2,16,16,16, 1,1,1,1, 0);
  launch_conv<3,1,1,1,128,128>(stream, hH + HOFF, p_tv, tvb, nullptr, vF + HOFF, nullptr, zp, 2,16,16,16, 2,16,16,16, 1,1,1,1, 0);
  // cross(xs, xe) and cross(xe, xs): 1024 windows each
  win_attn_kernel<<<1024, 64, 0, stream>>>(qF,        kF + HOFF, vF + HOFF, yH);
  win_attn_kernel<<<1024, 64, 0, stream>>>(qF + HOFF, kF,        vF,        yH + HOFF);
  // projection conv per half + residual add -> bufB
  launch_conv<3,1,1,1,128,128>(stream, yH,        p_tp, tpb, bufA,        bufB,        nullptr, zp, 2,16,16,16, 2,16,16,16, 1,1,1,1, 0);
  launch_conv<3,1,1,1,128,128>(stream, yH + HOFF, p_tp, tpb, bufA + HOFF, bufB + HOFF, nullptr, zp, 2,16,16,16, 2,16,16,16, 1,1,1,1, 0);

  // ================= Stage 2: slg (global attention over coarse+fine kv) ====
  cnorm(bufB, n2w, n2b, nullptr, hH, NTOK);
  // q conv (1,3,3,3), full grid
  launch_conv<1,3,3,3,128,128>(stream, hH, p_sq, sqb, nullptr, qF, nullptr, zp, 4,16,16,16, 4,16,16,16, 1,1,1,1, 0);
  // coarse: stride (4,8,8,8) -> (1,2,2,2); fine: stride (2,2,2,2) -> (2,8,8,8)
  launch_conv<1,3,3,3,128,128>(stream, hH, p_sc, scb, nullptr, nullptr, xcH, zp, 4,16,16,16, 1,2,2,2, 4,8,8,8, 0);
  launch_conv<1,3,3,3,128,128>(stream, hH, p_sf, sfb, nullptr, nullptr, xfH, zp, 4,16,16,16, 2,8,8,8, 2,2,2,2, 0);
  // k/v convs on downsampled grids -> concat [8 | 1024] kv rows
  launch_conv<1,3,3,3,128,128>(stream, xcH, p_sk, skb, nullptr, kvkF,           nullptr, zp, 1,2,2,2, 1,2,2,2, 1,1,1,1, 0);
  launch_conv<1,3,3,3,128,128>(stream, xfH, p_sk, skb, nullptr, kvkF + 8 * 128, nullptr, zp, 2,8,8,8, 2,8,8,8, 1,1,1,1, 0);
  launch_conv<1,3,3,3,128,128>(stream, xcH, p_sv, svb, nullptr, kvvF,           nullptr, zp, 1,2,2,2, 1,2,2,2, 1,1,1,1, 0);
  launch_conv<1,3,3,3,128,128>(stream, xfH, p_sv, svb, nullptr, kvvF + 8 * 128, nullptr, zp, 2,8,8,8, 2,8,8,8, 1,1,1,1, 0);
  // shared layernorm on concatenated kv
  cnorm(kvkF, snw, snb, nullptr, knH, NKV);
  cnorm(kvvF, snw, snb, nullptr, vnH, NKV);
  // global attention: 16384 q x 1032 kv x 8 heads
  global_attn_kernel<<<dim3(64, 8), 256, 0, stream>>>(qF, knH, vnH, yH, NKV);
  // projection conv + residual -> bufC
  launch_conv<1,3,3,3,128,128>(stream, yH, p_sp, spb, bufB, bufC, nullptr, zp, 4,16,16,16, 4,16,16,16, 1,1,1,1, 0);

  // ================= Stage 3: MLP (3^4 convs, 128->512 relu ->128) ==========
  cnorm(bufC, n3w, n3b, nullptr, hH, NTOK);
  launch_conv<3,3,3,3,128,512>(stream, hH,   p_m1, m1b, nullptr, nullptr, midH, zp, 4,16,16,16, 4,16,16,16, 1,1,1,1, 1);
  launch_conv<3,3,3,3,512,128>(stream, midH, p_m2, m2b, bufC,    bufA,    nullptr, zp, 4,16,16,16, 4,16,16,16, 1,1,1,1, 0);

  // ---- token-major -> (B,L,C,D,H,W) output ----
  tok_to_x_kernel<<<(int)((NF + 255) / 256), 256, 0, stream>>>(bufA, (float*)d_out);
}
